// MHSAWithMultiHeadRelativePositionEmbedding_4234837753954
// MI455X (gfx1250) — matrix operand order repaired and checked by reading, not compiled, not run on hardware
//
#include <hip/hip_runtime.h>

// ---------------------------------------------------------------------------
// MHSA with relative position bias for MI455X (gfx1250), all GEMM stages on
// v_wmma_f32_16x16x32_bf16. Wave32. GEMMs register-blocked 32x64 per wave
// (8 accumulator tiles, ~21 FLOP/byte from L2); K loop fully unrolled so the
// scheduler pipelines loads across chunks with no buffer-rotation copies.
// ---------------------------------------------------------------------------

typedef __bf16 v16bf __attribute__((ext_vector_type(16)));
typedef __bf16 v8bf  __attribute__((ext_vector_type(8)));
typedef float  v8f   __attribute__((ext_vector_type(8)));

#define BATCH   16
#define HDIM    28
#define WDIM    28
#define NTOK    784      // 28*28
#define NPAD    800      // 25 * 32  (K padding for P@V)
#define CH      512
#define F3      1536
#define HEADS   8
#define KD      64
#define NRD     3025     // 55*55
#define ROWS    12544    // BATCH * NTOK

static __device__ __forceinline__ v8f wmma_bf16(v16bf a, v16bf b, v8f c) {
  return __builtin_amdgcn_wmma_f32_16x16x32_bf16(false, a, false, b, (short)0, c,
                                                 false, false);
}

// Build a 16x32 A-operand register image for one lane from a row pointer.
// Lane layout (contiguous): lanes 0-15 hold K 0..15, lanes 16-31 hold
// K 16..31 of row (lane&15).
static __device__ __forceinline__ v16bf load_a(const __bf16* p, int half) {
  v8bf lo = *(const v8bf*)(p + half * 16);
  v8bf hi = *(const v8bf*)(p + half * 16 + 8);
  v16bf a;
#pragma unroll
  for (int j = 0; j < 8; ++j) { a[j] = lo[j]; a[8 + j] = hi[j]; }
  return a;
}

// ------------------------- elementwise conversions -------------------------

__global__ void f32_to_bf16_kernel(const float* __restrict__ in,
                                   __bf16* __restrict__ out, int n) {
  int i = blockIdx.x * blockDim.x + threadIdx.x;
  if (i < n) out[i] = (__bf16)in[i];
}

// out[c*rows + r] = in[r*cols + c]  (out is [cols][rows], coalesced on out)
__global__ void transpose_to_bf16_kernel(const float* __restrict__ in,
                                         __bf16* __restrict__ out,
                                         int rows, int cols) {
  int i = blockIdx.x * blockDim.x + threadIdx.x;
  if (i >= rows * cols) return;
  int c = i / rows;
  int r = i - c * rows;
  out[i] = (__bf16)in[r * cols + c];
}

// ----------------------------- QKV projection ------------------------------
// A = x_bf16 [ROWS][CH], Bt = wqkv transposed [F3][CH].
// grid (F3/64, ROWS/32), 32 threads; each wave computes a 32x64 strip.

__global__ __launch_bounds__(32)
void qkv_gemm_kernel(const __bf16* __restrict__ xb, const __bf16* __restrict__ wt,
                     __bf16* __restrict__ qg, __bf16* __restrict__ kg,
                     __bf16* __restrict__ vtg) {
  const int lane = threadIdx.x & 31;
  const int r    = lane & 15;       // A row within tile / B column
  const int half = lane >> 4;       // K half selector
  const int row0 = blockIdx.y * 32;
  const int f0   = blockIdx.x * 64; // 64-aligned: same (s,h) for whole strip

  const __bf16* ap0 = xb + (size_t)(row0 + r) * CH;
  const __bf16* ap1 = ap0 + (size_t)16 * CH;
  const __bf16* bp  = wt + (size_t)(f0 + r) * CH + half * 16;

  v8f acc[2][4] = {};
#pragma unroll
  for (int k0 = 0; k0 < CH; k0 += 32) {
    v16bf a0 = load_a(ap0 + k0, half);
    v16bf a1 = load_a(ap1 + k0, half);
    v16bf b[4];
#pragma unroll
    for (int ct = 0; ct < 4; ++ct)
      b[ct] = *(const v16bf*)(bp + (size_t)ct * 16 * CH + k0);
#pragma unroll
    for (int ct = 0; ct < 4; ++ct) {
      acc[0][ct] = wmma_bf16(a0, b[ct], acc[0][ct]);
      acc[1][ct] = wmma_bf16(a1, b[ct], acc[1][ct]);
    }
  }

  // f = s*512 + h*64 + d; strip is 64-aligned so s,h uniform, d = ct*16 + r.
  const int s  = f0 >> 9;
  const int h  = (f0 >> 6) & 7;
  const int b0 = row0 / NTOK;        // uniform
  const int n0 = row0 - b0 * NTOK;   // uniform; strip crosses <=1 batch boundary
  if (s == 0) {
#pragma unroll
    for (int rt = 0; rt < 2; ++rt)
#pragma unroll
      for (int i = 0; i < 8; ++i) {
        int nI = n0 + rt * 16 + i + half * 8;
        int bI = b0;
        if (nI >= NTOK) { nI -= NTOK; bI += 1; }
        size_t base = (((size_t)(bI * HEADS + h)) * NTOK + nI) * KD + r;
#pragma unroll
        for (int ct = 0; ct < 4; ++ct)
          qg[base + ct * 16] = (__bf16)(acc[rt][ct][i] * 0.125f);  // 1/sqrt(64)
      }
  } else if (s == 1) {
#pragma unroll
    for (int rt = 0; rt < 2; ++rt)
#pragma unroll
      for (int i = 0; i < 8; ++i) {
        int nI = n0 + rt * 16 + i + half * 8;
        int bI = b0;
        if (nI >= NTOK) { nI -= NTOK; bI += 1; }
        size_t base = (((size_t)(bI * HEADS + h)) * NTOK + nI) * KD + r;
#pragma unroll
        for (int ct = 0; ct < 4; ++ct)
          kg[base + ct * 16] = (__bf16)acc[rt][ct][i];
      }
  } else {
    // V stored transposed: vt[(bh*KD + d)*NPAD + nI]
#pragma unroll
    for (int rt = 0; rt < 2; ++rt)
#pragma unroll
      for (int i = 0; i < 8; ++i) {
        int nI = n0 + rt * 16 + i + half * 8;
        int bI = b0;
        if (nI >= NTOK) { nI -= NTOK; bI += 1; }
        size_t base = ((size_t)(bI * HEADS + h)) * KD * NPAD + nI;
#pragma unroll
        for (int ct = 0; ct < 4; ++ct)
          vtg[base + (size_t)(ct * 16 + r) * NPAD] = (__bf16)acc[rt][ct][i];
      }
  }
}

// ------------------------------- attention ---------------------------------
// One wave per (query block of 16, head, batch). Scores in LDS f32[16][800].

__global__ __launch_bounds__(32)
void attn_kernel(const __bf16* __restrict__ qg, const __bf16* __restrict__ kg,
                 const __bf16* __restrict__ vtg, const float* __restrict__ pos,
                 __bf16* __restrict__ ao) {
  __shared__ float sc[16][NPAD];
  __shared__ float red[32];
  __shared__ float rowstat[16];

  const int lane = threadIdx.x & 31;
  const int r    = lane & 15;
  const int half = lane >> 4;
  const int qblk = blockIdx.x;
  const int h    = blockIdx.y;
  const int b    = blockIdx.z;
  const int bh   = b * HEADS + h;
  const int q0   = qblk * 16;

  // Q tile as two A operands (K chunks 0..31 and 32..63)
  const __bf16* qp = qg + ((size_t)bh * NTOK + q0 + r) * KD;
  v16bf a0 = load_a(qp, half);
  v16bf a1 = load_a(qp + 32, half);

  // ---- scores = (Q/sqrt(d)) @ K^T + pos_bias ----
  for (int kt = 0; kt < 49; ++kt) {
    const __bf16* kp = kg + ((size_t)bh * NTOK + kt * 16 + r) * KD + half * 16;
    v16bf b0 = *(const v16bf*)(kp);
    v16bf b1 = *(const v16bf*)(kp + 32);
    v8f s = {};
    s = wmma_bf16(a0, b0, s);
    s = wmma_bf16(a1, b1, s);

    const int key = kt * 16 + r;
    const int kx  = key % HDIM;
    const int ky  = key / HDIM;
#pragma unroll
    for (int i = 0; i < 8; ++i) {
      int m  = i + half * 8;
      int qi = q0 + m;
      int r0 = (qi % HDIM) - kx + (HDIM - 1);
      int r1 = (qi / HDIM) - ky + (WDIM - 1);
      sc[m][key] = s[i] + pos[h * NRD + r1 * (2 * HDIM - 1) + r0];
    }
  }
  if (lane < 16) {
#pragma unroll
    for (int j = 0; j < 16; ++j) sc[lane][NTOK + j] = -1e30f;  // exp -> 0
  }
  __syncthreads();

  // ---- softmax (2 lanes per row: 400 columns each) ----
  float lmax = -1e30f;
  for (int c = half * 400; c < half * 400 + 400; ++c)
    lmax = fmaxf(lmax, sc[r][c]);
  red[lane] = lmax;
  __syncthreads();
  if (lane < 16) rowstat[lane] = fmaxf(red[lane], red[lane + 16]);
  __syncthreads();
  const float mx = rowstat[r];

  float lsum = 0.f;
  for (int c = half * 400; c < half * 400 + 400; ++c) {
    float e = __expf(sc[r][c] - mx);
    sc[r][c] = e;            // store un-normalized probs
    lsum += e;
  }
  red[lane] = lsum;
  __syncthreads();
  if (lane < 16) rowstat[lane] = red[lane] + red[lane + 16];
  __syncthreads();
  float invs[8];
#pragma unroll
  for (int i = 0; i < 8; ++i) invs[i] = 1.0f / rowstat[i + half * 8];

  // ---- out = P @ V  (P from LDS f32 -> bf16 in regs; V transposed) ----
  v8f o0 = {}, o1 = {}, o2 = {}, o3 = {};
  const __bf16* vbase = vtg + (size_t)bh * KD * NPAD;
  for (int kc = 0; kc < 25; ++kc) {
    const float* sp = &sc[r][kc * 32];
    v16bf pa;
#pragma unroll
    for (int j = 0; j < 8; ++j) {
      pa[j]     = (__bf16)sp[half * 16 + j];
      pa[8 + j] = (__bf16)sp[half * 16 + 8 + j];
    }
    const __bf16* vp = vbase + (size_t)r * NPAD + kc * 32 + half * 16;
    v16bf pb0 = *(const v16bf*)(vp);
    v16bf pb1 = *(const v16bf*)(vp + 16 * NPAD);
    v16bf pb2 = *(const v16bf*)(vp + 32 * NPAD);
    v16bf pb3 = *(const v16bf*)(vp + 48 * NPAD);
    o0 = wmma_bf16(pa, pb0, o0);
    o1 = wmma_bf16(pa, pb1, o1);
    o2 = wmma_bf16(pa, pb2, o2);
    o3 = wmma_bf16(pa, pb3, o3);
  }

  const size_t obase = ((size_t)b * NTOK + q0) * CH + h * KD;
#pragma unroll
  for (int i = 0; i < 8; ++i) {
    int m = i + half * 8;
    size_t off = obase + (size_t)m * CH + r;
    ao[off]      = (__bf16)(o0[i] * invs[i]);
    ao[off + 16] = (__bf16)(o1[i] * invs[i]);
    ao[off + 32] = (__bf16)(o2[i] * invs[i]);
    ao[off + 48] = (__bf16)(o3[i] * invs[i]);
  }
}

// ----------------------------- output projection ---------------------------
// grid (CH/64, ROWS/32); each wave computes a 32x64 strip, f32 to d_out.

__global__ __launch_bounds__(32)
void outproj_kernel(const __bf16* __restrict__ ao, const __bf16* __restrict__ wt,
                    float* __restrict__ out) {
  const int lane = threadIdx.x & 31;
  const int r    = lane & 15;
  const int half = lane >> 4;
  const int row0 = blockIdx.y * 32;
  const int c0   = blockIdx.x * 64;

  const __bf16* ap0 = ao + (size_t)(row0 + r) * CH;
  const __bf16* ap1 = ap0 + (size_t)16 * CH;
  const __bf16* bp  = wt + (size_t)(c0 + r) * CH + half * 16;

  v8f acc[2][4] = {};
#pragma unroll
  for (int k0 = 0; k0 < CH; k0 += 32) {
    v16bf a0 = load_a(ap0 + k0, half);
    v16bf a1 = load_a(ap1 + k0, half);
    v16bf b[4];
#pragma unroll
    for (int ct = 0; ct < 4; ++ct)
      b[ct] = *(const v16bf*)(bp + (size_t)ct * 16 * CH + k0);
#pragma unroll
    for (int ct = 0; ct < 4; ++ct) {
      acc[0][ct] = wmma_bf16(a0, b[ct], acc[0][ct]);
      acc[1][ct] = wmma_bf16(a1, b[ct], acc[1][ct]);
    }
  }

#pragma unroll
  for (int rt = 0; rt < 2; ++rt)
#pragma unroll
    for (int i = 0; i < 8; ++i) {
      size_t rowoff = (size_t)(row0 + rt * 16 + i + half * 8) * CH + c0 + r;
#pragma unroll
      for (int ct = 0; ct < 4; ++ct)
        out[rowoff + ct * 16] = acc[rt][ct][i];
    }
}

// --------------------------------- launch ----------------------------------

extern "C" void kernel_launch(void* const* d_in, const int* in_sizes, int n_in,
                              void* d_out, int out_size, void* d_ws, size_t ws_size,
                              hipStream_t stream) {
  (void)in_sizes; (void)n_in; (void)out_size; (void)ws_size;
  const float* x      = (const float*)d_in[0];
  const float* w_qkv  = (const float*)d_in[1];
  const float* postab = (const float*)d_in[2];
  const float* w_out  = (const float*)d_in[3];
  float* out = (float*)d_out;

  char* p = (char*)d_ws;
  auto alloc = [&](size_t bytes) -> void* {
    void* r = (void*)p;
    p += (bytes + 255) & ~(size_t)255;
    return r;
  };
  const size_t vt_bytes = (size_t)BATCH * HEADS * KD * NPAD * 2;
  __bf16* xb  = (__bf16*)alloc((size_t)ROWS * CH * 2);
  __bf16* wqt = (__bf16*)alloc((size_t)F3 * CH * 2);
  __bf16* wot = (__bf16*)alloc((size_t)CH * CH * 2);
  __bf16* qb  = (__bf16*)alloc((size_t)BATCH * HEADS * NTOK * KD * 2);
  __bf16* kb  = (__bf16*)alloc((size_t)BATCH * HEADS * NTOK * KD * 2);
  __bf16* vtb = (__bf16*)alloc(vt_bytes);
  __bf16* ao  = (__bf16*)alloc((size_t)ROWS * CH * 2);

  const int nx = ROWS * CH;
  f32_to_bf16_kernel<<<(nx + 255) / 256, 256, 0, stream>>>(x, xb, nx);
  transpose_to_bf16_kernel<<<(CH * F3 + 255) / 256, 256, 0, stream>>>(w_qkv, wqt, CH, F3);
  transpose_to_bf16_kernel<<<(CH * CH + 255) / 256, 256, 0, stream>>>(w_out, wot, CH, CH);
  hipMemsetAsync(vtb, 0, vt_bytes, stream);  // zero the N-padding of V^T

  qkv_gemm_kernel<<<dim3(F3 / 64, ROWS / 32), 32, 0, stream>>>(xb, wqt, qb, kb, vtb);
  attn_kernel<<<dim3(NTOK / 16, HEADS, BATCH), 32, 0, stream>>>(qb, kb, vtb, postab, ao);
  outproj_kernel<<<dim3(CH / 64, ROWS / 32), 32, 0, stream>>>(ao, wot, out);
}
